// StochasticPool2d_78847009620558
// MI455X (gfx1250) — matrix operand order repaired and checked by reading, not compiled, not run on hardware
//
#include <hip/hip_runtime.h>
#include <stdint.h>

// Problem constants (x: [8, 64, 128, 128] fp32; k=3, stride=1, pad=1 -> Ho=Wo=128)
#define B_    8
#define C_    64
#define H_    128
#define W_    128
#define HW_   (H_ * W_)
#define TILE  16
#define XT_R  18      // 16 + 1 halo each side (window tile needs rows/cols -1..16)
#define XT_S  19      // odd fp32 LDS stride -> bank-conflict free
#define YV_S  17      // stride for 18x16 horizontal-sum array
#define SM_S  17      // stride for 16x16 window-sum array
#define ST_S  20      // byte stride for sel tile in pass 2
#define EPS   1e-6f

typedef __attribute__((ext_vector_type(16))) _Float16 v16h;
typedef __attribute__((ext_vector_type(8)))  float    v8f;

// ---- deterministic counter-based RNG (one draw per window) -----------------
__device__ __forceinline__ uint32_t mix32(uint32_t x) {
    x ^= x >> 17; x *= 0xED5AD4BBu;
    x ^= x >> 11; x *= 0xAC4C1B51u;
    x ^= x >> 15; x *= 0x31848BABu;
    x ^= x >> 14;
    return x;
}

// ---- CDNA5 async global->LDS copies (ASYNCcnt-tracked) ---------------------
__device__ __forceinline__ void async_g2l_b32(uint32_t lds_off, uint64_t gaddr) {
    asm volatile("global_load_async_to_lds_b32 %0, %1, off"
                 :: "v"(lds_off), "v"(gaddr) : "memory");
}
__device__ __forceinline__ void async_g2l_b8(uint32_t lds_off, uint64_t gaddr) {
    asm volatile("global_load_async_to_lds_b8 %0, %1, off"
                 :: "v"(lds_off), "v"(gaddr) : "memory");
}
__device__ __forceinline__ void wait_async0() {
#if __has_builtin(__builtin_amdgcn_s_wait_asynccnt)
    __builtin_amdgcn_s_wait_asynccnt(0);
#else
    asm volatile("s_wait_asynccnt 0x0" ::: "memory");
#endif
}

// ============================================================================
// Pass 1: per-window categorical sample -> sel byte in d_ws.
//   Window sums via separable box filter; the vertical 3x1 reduction is one
//   v_wmma_f32_16x16x32_f16:  S(16x16) = T(16x32, banded 0/1) x Yv(32x16).
// ============================================================================
__global__ __launch_bounds__(256)
void sp_pass1_sel(const float* __restrict__ x, unsigned char* __restrict__ sel) {
    __shared__ float xt[XT_R * XT_S];   // 18x18 input tile, origin (h0-1, w0-1)
    __shared__ float yv[XT_R * YV_S];   // yv[r][c] = horizontal 1x3 sum, r:0..17, c:0..15
    __shared__ float sm[TILE * SM_S];   // 16x16 window sums (WMMA D matrix)

    const int tx    = threadIdx.x;
    const int ty    = threadIdx.y;
    const int tid   = ty * TILE + tx;   // 8 wave32s
    const int plane = blockIdx.z;
    const int h0    = blockIdx.y * TILE;
    const int w0    = blockIdx.x * TILE;
    const float* __restrict__ xp = x + (size_t)plane * HW_;

    // zero-fill tile (zero padding at image border)
#pragma unroll
    for (int rr = 0; rr < 2; ++rr) {
        const int i = tid + rr * 256;
        if (i < XT_R * XT_S) xt[i] = 0.0f;
    }
    __syncthreads();

    // async-stage 18x18 halo, global -> LDS
#pragma unroll
    for (int rr = 0; rr < 2; ++rr) {
        const int i = tid + rr * 256;
        if (i < XT_R * XT_R) {
            const int r = i / XT_R, c = i % XT_R;
            const int gh = h0 - 1 + r, gw = w0 - 1 + c;
            if (gh >= 0 && gh < H_ && gw >= 0 && gw < W_)
                async_g2l_b32((uint32_t)(uintptr_t)&xt[r * XT_S + c],
                              (uint64_t)(uintptr_t)(xp + gh * W_ + gw));
        }
    }
    wait_async0();
    __syncthreads();

    // horizontal 1x3 box sums (VALU, contiguous)
#pragma unroll
    for (int rr = 0; rr < 2; ++rr) {
        const int i = tid + rr * 256;
        if (i < XT_R * TILE) {
            const int r = i >> 4, c = i & 15;
            yv[r * YV_S + c] =
                xt[r * XT_S + c] + xt[r * XT_S + c + 1] + xt[r * XT_S + c + 2];
        }
    }
    __syncthreads();

    // vertical 3x1 reduction on the matrix pipe: S = T x Yv  (K=18 <= 32)
    {
        const int lane = tid & 31;      // wave32 lane
        const int half = lane >> 4;     // 0: lanes 0-15, 1: lanes 16-31
        const int nrow = lane & 15;     // A: M row; B: N column
        v16h a, b;
#pragma unroll
        for (int v = 0; v < 8; ++v) {
#pragma unroll
            for (int hh = 0; hh < 2; ++hh) {
                // A (16x32 f16): K = 0-15 in lanes 0-15, K+16 in lanes 16-31
                const int kA = 2 * v + hh + half * 16;
                a[2 * v + hh] = (_Float16)((kA >= nrow && kA <= nrow + 2) ? 1.0f : 0.0f);
                // B (32x16 f16): K pairs per VGPR; lanes 16-31 carry K+16
                const int kB = 2 * v + hh + half * 16;
                b[2 * v + hh] = (_Float16)((kB < XT_R) ? yv[kB * YV_S + nrow] : 0.0f);
            }
        }
        v8f c = {};
        c = __builtin_amdgcn_wmma_f32_16x16x32_f16(
                /*neg_a=*/false, a, /*neg_b=*/false, b,
                /*c_mod=*/(short)0, c, /*reuse_a=*/false, /*reuse_b=*/false);
        if (tid < 32) {                 // wave-uniform: wave 0 publishes S
#pragma unroll
            for (int v = 0; v < 8; ++v)
                sm[(v + 8 * half) * SM_S + nrow] = c[v];  // D: VGPR v -> M = v + 8*half
        }
    }
    __syncthreads();

    // inverse-CDF sample for window (ho, wo) = (h0+ty, w0+tx)
    float v9[9];
#pragma unroll
    for (int i = 0; i < 3; ++i)
#pragma unroll
        for (int j = 0; j < 3; ++j)
            v9[i * 3 + j] = xt[(ty + i) * XT_S + (tx + j)];

    const float s   = sm[ty * SM_S + tx];
    const int   ho  = h0 + ty, wo = w0 + tx;
    const uint32_t widx = (uint32_t)(plane * HW_ + ho * W_ + wo);
    const float u = (float)(mix32(widx ^ (42u * 0x9E3779B9u)) >> 8) * (1.0f / 16777216.0f);

    const float thresh = u * (s + EPS);
    float acc = 0.0f;
    int   selk = 8;
#pragma unroll
    for (int k = 0; k < 9; ++k) {
        acc += v9[k];
        if (selk == 8 && acc > thresh) selk = k;
    }
    sel[(size_t)plane * HW_ + ho * W_ + wo] = (unsigned char)selk;
}

// ============================================================================
// Pass 2: gather. out[p] = x[p] * (#covering windows that sampled p) / coverage.
//   sel halo staged via async B8 loads; 0xFF fill makes border handling branchless.
// ============================================================================
__global__ __launch_bounds__(256)
void sp_pass2_gather(const float* __restrict__ x,
                     const unsigned char* __restrict__ sel,
                     float* __restrict__ out) {
    __shared__ uint32_t st_u32[(XT_R * ST_S) / 4];          // 360 B, 4B-aligned
    unsigned char* st = (unsigned char*)st_u32;

    const int tx    = threadIdx.x;
    const int ty    = threadIdx.y;
    const int tid   = ty * TILE + tx;
    const int plane = blockIdx.z;
    const int h0    = blockIdx.y * TILE;
    const int w0    = blockIdx.x * TILE;

    if (tid < (XT_R * ST_S) / 4) st_u32[tid] = 0xFFFFFFFFu;  // 0xFF = no window
    __syncthreads();

#pragma unroll
    for (int rr = 0; rr < 2; ++rr) {
        const int i = tid + rr * 256;
        if (i < XT_R * XT_R) {
            const int r = i / XT_R, c = i % XT_R;
            const int gh = h0 - 1 + r, gw = w0 - 1 + c;
            if (gh >= 0 && gh < H_ && gw >= 0 && gw < W_)
                async_g2l_b8((uint32_t)(uintptr_t)&st[r * ST_S + c],
                             (uint64_t)(uintptr_t)(sel + (size_t)plane * HW_ + gh * W_ + gw));
        }
    }
    wait_async0();
    __syncthreads();

    const int h = h0 + ty, w = w0 + tx;
    float m = 0.0f;
#pragma unroll
    for (int dh = -1; dh <= 1; ++dh)
#pragma unroll
        for (int dw = -1; dw <= 1; ++dw) {
            const int sv = st[(ty + 1 + dh) * ST_S + (tx + 1 + dw)];
            m += (sv == (1 - dh) * 3 + (1 - dw)) ? 1.0f : 0.0f;
        }

    const float cnt = (float)((3 - (h == 0) - (h == H_ - 1)) *
                              (3 - (w == 0) - (w == W_ - 1)));
    const float xv = x[(size_t)plane * HW_ + h * W_ + w];
    __builtin_nontemporal_store(xv * m / cnt,
                                out + (size_t)plane * HW_ + h * W_ + w);
}

extern "C" void kernel_launch(void* const* d_in, const int* in_sizes, int n_in,
                              void* d_out, int out_size, void* d_ws, size_t ws_size,
                              hipStream_t stream) {
    (void)in_sizes; (void)n_in; (void)out_size; (void)ws_size;
    const float*   x   = (const float*)d_in[0];
    float*         out = (float*)d_out;
    unsigned char* sel = (unsigned char*)d_ws;   // 8*64*128*128 = 8.4 MB sel bytes

    dim3 block(TILE, TILE, 1);                   // 256 threads = 8 wave32s
    dim3 grid(W_ / TILE, H_ / TILE, B_ * C_);    // 8 x 8 x 512

    sp_pass1_sel<<<grid, block, 0, stream>>>(x, sel);
    sp_pass2_gather<<<grid, block, 0, stream>>>(x, sel, out);
}
